// SupervisedGraphsage_65034394796834
// MI455X (gfx1250) — compile-verified, matrix operand order checked by reading, not run on hardware
//
#include <hip/hip_runtime.h>
#include <hip/hip_bf16.h>

// ---------------------------------------------------------------------------
// GraphSAGE forward for MI455X (gfx1250, wave32, WMMA).
// Gathers/means are bandwidth-bound coalesced f32 streams; all GEMMs use
// v_wmma_f32_16x16x32_f16 (f16 inputs, f32 accumulate), one 16x16 tile/wave.
// ---------------------------------------------------------------------------

typedef __attribute__((ext_vector_type(16))) _Float16 v16h;
typedef __attribute__((ext_vector_type(8)))  _Float16 v8h;
typedef __attribute__((ext_vector_type(8)))  float    v8f;

#define D_FEAT 128
#define FAN0   25
#define FAN1   10

// --- Convert+transpose fp32 weight [inDim, outDim] -> f16 Wt [outPad, inDim].
// Rows of Wt are contiguous K so WMMA B-fragment loads are plain B128 loads.
__global__ void k_w2h(const float* __restrict__ W, _Float16* __restrict__ Wt,
                      int inDim, int outDim, int outPad) {
    int idx = blockIdx.x * blockDim.x + threadIdx.x;
    int total = inDim * outPad;
    if (idx >= total) return;
    int o = idx / inDim;
    int i = idx - o * inDim;
    float v = (o < outDim) ? W[(size_t)i * outDim + o] : 0.0f;
    Wt[(size_t)o * inDim + i] = (_Float16)v;
}

// --- Gather h0, h1 and mean m1 (one block per batch node, 128 threads = cols).
__global__ void k_gather0(const float* __restrict__ feat,
                          const int* __restrict__ adj,
                          const long long* __restrict__ batch,
                          _Float16* __restrict__ h0h,
                          _Float16* __restrict__ h1h,
                          _Float16* __restrict__ m1h,
                          int* __restrict__ s1buf) {
    int b = blockIdx.x, c = threadIdx.x;
    long long s0 = batch[b];
    float f0 = feat[(size_t)s0 * D_FEAT + c];
    h0h[(size_t)b * D_FEAT + c] = (_Float16)f0;
    const int* arow = adj + (size_t)s0 * 128;
    float acc = 0.0f;
#pragma unroll
    for (int j = 0; j < FAN1; ++j) {
        int s1i = arow[j];                    // uniform across block -> scalar
        if (c == 0) s1buf[b * FAN1 + j] = s1i;
        float v = feat[(size_t)s1i * D_FEAT + c];
        h1h[((size_t)(b * FAN1 + j)) * D_FEAT + c] = (_Float16)v;
        acc += v;
    }
    m1h[(size_t)b * D_FEAT + c] = (_Float16)(acc * (1.0f / FAN1));
}

// --- Gather 25 neighbor rows and mean -> m2 (one block per s1 node).
__global__ void k_gather1(const float* __restrict__ feat,
                          const int* __restrict__ adj,
                          const int* __restrict__ s1buf,
                          _Float16* __restrict__ m2h) {
    int i = blockIdx.x, c = threadIdx.x;
    int s = s1buf[i];
    const int* arow = adj + (size_t)s * 128;
    float acc = 0.0f;
#pragma unroll 5
    for (int j = 0; j < FAN0; ++j) {
        int idx = arow[j];                    // uniform -> scalar load
        acc += feat[(size_t)idx * D_FEAT + c];
    }
    m2h[(size_t)i * D_FEAT + c] = (_Float16)(acc * (1.0f / FAN0));
}

// --- WMMA GEMM: Out[m, colOff+n] = act(sum_k A[m,k] * Wt[n,k]).
// One wave (32 threads) per 16x16 tile; K stepped by 32.
// A: f16 row-major [M, lda]; Wt: f16 row-major [N, K] (pre-transposed weight).
template <bool RELU, bool OUTF16>
__global__ void k_wmma_gemm(const _Float16* __restrict__ A, int lda,
                            const _Float16* __restrict__ Wt, int K,
                            _Float16* __restrict__ outH,
                            float* __restrict__ outF,
                            int ldOut, int colOff, int nLimit) {
    const int lane  = threadIdx.x;            // 0..31 (wave32)
    const int tm    = blockIdx.x;
    const int tn    = blockIdx.y;
    const int l15   = lane & 15;
    const int kbase = (lane >> 4) << 3;       // 0 for lanes 0-15, 8 for 16-31

    const _Float16* arow = A  + (size_t)(tm * 16 + l15) * lda;
    const _Float16* brow = Wt + (size_t)(tn * 16 + l15) * K;

    v8f c = {};
    for (int k0 = 0; k0 < K; k0 += 32) {
        // 16-bit A layout: lane<16 holds K {0..7,16..23}, lane>=16 {8..15,24..31}
        v8h a0 = *(const v8h*)(arow + k0 + kbase);
        v8h a1 = *(const v8h*)(arow + k0 + 16 + kbase);
        v8h b0 = *(const v8h*)(brow + k0 + kbase);
        v8h b1 = *(const v8h*)(brow + k0 + 16 + kbase);
        v16h a, b;
#pragma unroll
        for (int t = 0; t < 8; ++t) {
            a[t] = a0[t]; a[t + 8] = a1[t];
            b[t] = b0[t]; b[t + 8] = b1[t];
        }
        c = __builtin_amdgcn_wmma_f32_16x16x32_f16(
                /*neg_a=*/false, a, /*neg_b=*/false, b,
                /*c_mod=*/(short)0, c, /*reuse_a=*/false, /*reuse_b=*/false);
    }

    int col = tn * 16 + l15;
    if (col >= nLimit) return;
    int rbase = tm * 16 + ((lane >> 4) << 3);  // C/D: lanes>=16 hold M=8..15
#pragma unroll
    for (int i = 0; i < 8; ++i) {
        float v = c[i];
        if (RELU) v = v > 0.0f ? v : 0.0f;
        size_t o = (size_t)(rbase + i) * ldOut + col + colOff;
        if (OUTF16) outH[o] = (_Float16)v;
        else        outF[o] = v;
    }
}

// --- Mean over groups of 10 rows of n1 [B*10, 256] -> mn1 [B, 256].
__global__ void k_mean_n1(const _Float16* __restrict__ n1h,
                          _Float16* __restrict__ mn1h) {
    int b = blockIdx.x, c = threadIdx.x;
    float acc = 0.0f;
#pragma unroll
    for (int j = 0; j < FAN1; ++j)
        acc += (float)n1h[((size_t)(b * FAN1 + j)) * 256 + c];
    mn1h[(size_t)b * 256 + c] = (_Float16)(acc * (1.0f / FAN1));
}

// --- Row-wise L2 normalize [B, 256] f32 -> f16 (LDS tree reduction).
__global__ void k_l2norm(const float* __restrict__ out0,
                         _Float16* __restrict__ outnh) {
    __shared__ float red[256];
    int b = blockIdx.x, c = threadIdx.x;
    float v = out0[(size_t)b * 256 + c];
    red[c] = v * v;
    __syncthreads();
#pragma unroll
    for (int s = 128; s > 0; s >>= 1) {
        if (c < s) red[c] += red[c + s];
        __syncthreads();
    }
    float scale = rsqrtf(fmaxf(red[0], 1e-12f));
    outnh[(size_t)b * 256 + c] = (_Float16)(v * scale);
}

extern "C" void kernel_launch(void* const* d_in, const int* in_sizes, int n_in,
                              void* d_out, int out_size, void* d_ws, size_t ws_size,
                              hipStream_t stream) {
    const float*      feat    = (const float*)d_in[0];      // [N, 128]
    const int*        adj     = (const int*)d_in[1];        // [N, 128]
    const long long*  batch   = (const long long*)d_in[2];  // [B] int64
    const float*      wself0  = (const float*)d_in[3];      // [128, 128]
    const float*      wneigh0 = (const float*)d_in[4];      // [128, 128]
    const float*      wself1  = (const float*)d_in[5];      // [256, 128]
    const float*      wneigh1 = (const float*)d_in[6];      // [256, 128]
    const float*      wpred   = (const float*)d_in[7];      // [256, 50]
    float*            out     = (float*)d_out;              // [B, 50]

    const int B  = in_sizes[2];        // 1024
    const int S1 = B * FAN1;           // 10240
    const int NC = 50, NCPAD = 64;

    // ---- carve workspace (256B-aligned regions) ----
    char* p = (char*)d_ws;
    auto alloc = [&](size_t bytes) {
        char* r = p;
        p += (bytes + 255) & ~(size_t)255;
        return r;
    };
    _Float16* wt_self0  = (_Float16*)alloc(128 * 128 * 2);
    _Float16* wt_neigh0 = (_Float16*)alloc(128 * 128 * 2);
    _Float16* wt_self1  = (_Float16*)alloc(128 * 256 * 2);
    _Float16* wt_neigh1 = (_Float16*)alloc(128 * 256 * 2);
    _Float16* wt_pred   = (_Float16*)alloc((size_t)NCPAD * 256 * 2);
    _Float16* h0h       = (_Float16*)alloc((size_t)B  * 128 * 2);
    _Float16* h1h       = (_Float16*)alloc((size_t)S1 * 128 * 2);
    _Float16* m1h       = (_Float16*)alloc((size_t)B  * 128 * 2);
    _Float16* m2h       = (_Float16*)alloc((size_t)S1 * 128 * 2);
    int*      s1buf     = (int*)alloc((size_t)S1 * 4);
    _Float16* n1h       = (_Float16*)alloc((size_t)S1 * 256 * 2);
    _Float16* n0h       = (_Float16*)alloc((size_t)B  * 256 * 2);
    _Float16* mn1h      = (_Float16*)alloc((size_t)B  * 256 * 2);
    float*    out0f     = (float*)alloc((size_t)B * 256 * 4);
    _Float16* outnh     = (_Float16*)alloc((size_t)B * 256 * 2);

    // ---- 0: weight convert + transpose to [N, K] f16 ----
    auto wlaunch = [&](const float* W, _Float16* Wt, int inD, int outD, int outP) {
        int total = inD * outP;
        k_w2h<<<(total + 255) / 256, 256, 0, stream>>>(W, Wt, inD, outD, outP);
    };
    wlaunch(wself0,  wt_self0,  128, 128, 128);
    wlaunch(wneigh0, wt_neigh0, 128, 128, 128);
    wlaunch(wself1,  wt_self1,  256, 128, 128);
    wlaunch(wneigh1, wt_neigh1, 256, 128, 128);
    wlaunch(wpred,   wt_pred,   256, NC,  NCPAD);

    // ---- 1: gathers + neighbor means ----
    k_gather0<<<B,  D_FEAT, 0, stream>>>(feat, adj, batch, h0h, h1h, m1h, s1buf);
    k_gather1<<<S1, D_FEAT, 0, stream>>>(feat, adj, s1buf, m2h);

    // ---- 2: layer 0 (ReLU, concat 128+128 -> 256, f16 out) ----
    dim3 gN1(S1 / 16, 8), gN0(B / 16, 8);
    k_wmma_gemm<true, true><<<gN1, 32, 0, stream>>>(h1h, 128, wt_self0,  128,
                                                    n1h, nullptr, 256, 0,   128);
    k_wmma_gemm<true, true><<<gN1, 32, 0, stream>>>(m2h, 128, wt_neigh0, 128,
                                                    n1h, nullptr, 256, 128, 128);
    k_wmma_gemm<true, true><<<gN0, 32, 0, stream>>>(h0h, 128, wt_self0,  128,
                                                    n0h, nullptr, 256, 0,   128);
    k_wmma_gemm<true, true><<<gN0, 32, 0, stream>>>(m1h, 128, wt_neigh0, 128,
                                                    n0h, nullptr, 256, 128, 128);

    // ---- 3: mean over n1 groups ----
    k_mean_n1<<<B, 256, 0, stream>>>(n1h, mn1h);

    // ---- 4: layer 1 (identity act, K=256, f32 out) ----
    k_wmma_gemm<false, false><<<gN0, 32, 0, stream>>>(n0h,  256, wt_self1,  256,
                                                      nullptr, out0f, 256, 0,   128);
    k_wmma_gemm<false, false><<<gN0, 32, 0, stream>>>(mn1h, 256, wt_neigh1, 256,
                                                      nullptr, out0f, 256, 128, 128);

    // ---- 5: row L2 normalize ----
    k_l2norm<<<B, 256, 0, stream>>>(out0f, outnh);

    // ---- 6: prediction GEMM [B,256] x [256,50] -> d_out ----
    dim3 gP(B / 16, NCPAD / 16);
    k_wmma_gemm<false, false><<<gP, 32, 0, stream>>>(outnh, 256, wt_pred, 256,
                                                     nullptr, out, NC, 0, NC);
}
